// Llama4MoE_20684562497547
// MI455X (gfx1250) — matrix operand order repaired and checked by reading, not compiled, not run on hardware
//
#include <hip/hip_runtime.h>
#include <hip/hip_bf16.h>

// ---------------------------------------------------------------------------
// Llama4-style MoE (T=4096, H=1024, I=2048, E=16, top_k=1) for gfx1250.
// fp32 weights streamed from HBM, converted to bf16 into LDS, all GEMMs via
// v_wmma_f32_16x16x32_bf16 (fp32 accumulate). Block tile 64x256, wave tile
// 32x64 (two A fragments share each B fragment).
// ---------------------------------------------------------------------------

#define T_TOK 4096
#define HDIM  1024
#define IDIM  2048
#define NEXP  16
#define ROWS_MAX 4416   // T + 16 experts * 15 pad + 80 slack (BM=64 overread)

#define BM 64
#define BN 256
#define BK 32
#define BKP 36          // padded LDS row stride (elements) to spread banks
#define NTHR 256

typedef __bf16 bf16;
typedef __attribute__((ext_vector_type(16))) __bf16 v16bf;
typedef __attribute__((ext_vector_type(8)))  float  v8f;

// fast sigmoid building block: v_rcp_f32 instead of precise fdiv
__device__ __forceinline__ float fast_rcp(float x) {
    return __builtin_amdgcn_rcpf(x);
}

// ---------------- WMMA fragment loaders (per CDNA5 ISA VGPR layouts) --------

// A 16x32 bf16: lane = m + 16*h; VGPR0-3 hold K=h*8..h*8+7, VGPR4-7 hold K=16+h*8..+7
__device__ __forceinline__ v16bf load_frag_A(const bf16* As, int lane) {
    int m = lane & 15, h = lane >> 4;
    const bf16* p = As + m * BKP + h * 8;
    union { v16bf v; uint2 d[4]; } f;
    f.d[0] = *(const uint2*)(p);
    f.d[1] = *(const uint2*)(p + 4);
    f.d[2] = *(const uint2*)(p + 16);
    f.d[3] = *(const uint2*)(p + 20);
    return f.v;
}

// B 32x16 bf16: lane = n + 16*kh; VGPR0-3 hold K=kh*8..kh*8+7, VGPR4-7 hold K=16+kh*8..+7
// LDS layout is fragment-major: Bs[n * BKP + k] -> per lane two runs of 8 bf16 (k=kh*8..+7, 16+kh*8..+7).
__device__ __forceinline__ v16bf load_frag_B(const bf16* Bs, int colOff, int lane) {
    int n  = colOff + (lane & 15);
    int kh = lane >> 4;
    const bf16* p = Bs + n * BKP + kh * 8;
    union { v16bf v; uint2 d[4]; } f;
    f.d[0] = *(const uint2*)(p);
    f.d[1] = *(const uint2*)(p + 4);
    f.d[2] = *(const uint2*)(p + 16);
    f.d[3] = *(const uint2*)(p + 20);
    return f.v;
}

// ---------------- cooperative tile loaders ----------------------------------

// A tile: BM(64) x BK(32) bf16 from global (already bf16), row-major copy.
__device__ __forceinline__ void load_A_tile(bf16* As, const bf16* A, int lda,
                                            int k0, int tid) {
    int r  = tid >> 2;          // 0..63
    int c8 = (tid & 3) * 8;     // 0,8,16,24
    uint4 v = *(const uint4*)(A + (size_t)r * lda + k0 + c8);   // 8 bf16
    uint2 lo, hi;
    lo.x = v.x; lo.y = v.y; hi.x = v.z; hi.y = v.w;
    *(uint2*)(As + r * BKP + c8)     = lo;
    *(uint2*)(As + r * BKP + c8 + 4) = hi;
}

// B tile: BK(32) x BN(256) fp32 from global, convert to bf16, store transposed
// (fragment-major) into LDS: Bs[n * BKP + k]. k is the contiguous LDS dim, so
// pack (k, k+1) bf16 pairs -> ds_store_b32 (v_cvt_pk_bf16_f32 path).
__device__ __forceinline__ void load_B_tile(bf16* Bs, const float* W, int ldw,
                                            int k0, int n0, int tid) {
    int n4 = tid & 63;          // float4 column index (64*4 = 256 cols)
    int kp = tid >> 6;          // 0..3 -> k-pair groups
    int nb = n4 * 4;
#pragma unroll
    for (int s = 0; s < 4; ++s) {
        int k = kp * 8 + s * 2;                       // even k
        const float* base = W + (size_t)(k0 + k) * ldw + n0 + n4 * 4;
        float4 f0 = *(const float4*)(base);           // row k
        float4 f1 = *(const float4*)(base + ldw);     // row k+1
        union { bf16 b[2]; unsigned u; } p0, p1, p2, p3;
        p0.b[0] = (bf16)f0.x; p0.b[1] = (bf16)f1.x;
        p1.b[0] = (bf16)f0.y; p1.b[1] = (bf16)f1.y;
        p2.b[0] = (bf16)f0.z; p2.b[1] = (bf16)f1.z;
        p3.b[0] = (bf16)f0.w; p3.b[1] = (bf16)f1.w;
        *(unsigned*)(Bs + (nb + 0) * BKP + k) = p0.u;
        *(unsigned*)(Bs + (nb + 1) * BKP + k) = p1.u;
        *(unsigned*)(Bs + (nb + 2) * BKP + k) = p2.u;
        *(unsigned*)(Bs + (nb + 3) * BKP + k) = p3.u;
    }
}

// ---------------- router: one wave per token --------------------------------

__global__ __launch_bounds__(256) void router_kernel(
    const float* __restrict__ x, const float* __restrict__ rw,
    int* __restrict__ eidx, float* __restrict__ scale, int* __restrict__ counts) {
    int wave = threadIdx.x >> 5, lane = threadIdx.x & 31;
    int t = blockIdx.x * 8 + wave;
    float acc[NEXP];
#pragma unroll
    for (int e = 0; e < NEXP; ++e) acc[e] = 0.f;
    for (int hi = 0; hi < HDIM / 32; ++hi) {
        int h = hi * 32 + lane;
        float xv = x[(size_t)t * HDIM + h];
        const float4* r4 = (const float4*)(rw + (size_t)h * NEXP);
        float4 a = r4[0], b = r4[1], c = r4[2], d = r4[3];
        acc[0]  += xv * a.x;  acc[1]  += xv * a.y;  acc[2]  += xv * a.z;  acc[3]  += xv * a.w;
        acc[4]  += xv * b.x;  acc[5]  += xv * b.y;  acc[6]  += xv * b.z;  acc[7]  += xv * b.w;
        acc[8]  += xv * c.x;  acc[9]  += xv * c.y;  acc[10] += xv * c.z;  acc[11] += xv * c.w;
        acc[12] += xv * d.x;  acc[13] += xv * d.y;  acc[14] += xv * d.z;  acc[15] += xv * d.w;
    }
#pragma unroll
    for (int off = 16; off > 0; off >>= 1)
#pragma unroll
        for (int e = 0; e < NEXP; ++e) acc[e] += __shfl_xor(acc[e], off, 32);
    if (lane == 0) {
        int bi = 0; float bv = acc[0];
#pragma unroll
        for (int e = 1; e < NEXP; ++e)
            if (acc[e] > bv) { bv = acc[e]; bi = e; }
        eidx[t]  = bi;
        scale[t] = fast_rcp(1.f + __expf(-bv));   // sigmoid
        atomicAdd(&counts[bi], 1);
    }
}

// ---------------- per-expert offsets (rows padded to 16) --------------------

__global__ void offsets_kernel(const int* __restrict__ counts, int* __restrict__ offsets) {
    if (threadIdx.x == 0 && blockIdx.x == 0) {
        int acc = 0;
        for (int e = 0; e < NEXP; ++e) {
            offsets[e] = acc;
            acc += (counts[e] + 15) & ~15;
        }
        offsets[NEXP] = acc;
    }
}

// ---------------- scatter: bucketize + fp32->bf16 (+ routing-weight scale) --

__global__ __launch_bounds__(256) void scatter_kernel(
    const float* __restrict__ x, const int* __restrict__ eidx,
    const float* __restrict__ scale, const int* __restrict__ offsets,
    int* __restrict__ cursor, int* __restrict__ perm,
    bf16* __restrict__ Xg, bf16* __restrict__ Xs) {
    int t = blockIdx.x;
    __shared__ int spos;
    if (threadIdx.x == 0) {
        int e = eidx[t];
        int p = offsets[e] + atomicAdd(&cursor[e], 1);
        perm[p] = t;
        spos = p;
    }
    __syncthreads();
    int pos = spos;
    float s = scale[t];
    int c4 = threadIdx.x * 4;   // 256 threads * 4 = 1024 = HDIM
    float4 v = *(const float4*)(x + (size_t)t * HDIM + c4);
    union { bf16 b[4]; uint2 u; } un, sc;
    un.b[0] = (bf16)v.x;       un.b[1] = (bf16)v.y;
    un.b[2] = (bf16)v.z;       un.b[3] = (bf16)v.w;
    sc.b[0] = (bf16)(v.x * s); sc.b[1] = (bf16)(v.y * s);
    sc.b[2] = (bf16)(v.z * s); sc.b[3] = (bf16)(v.w * s);
    *(uint2*)(Xs + (size_t)t   * HDIM + c4) = un.u;
    *(uint2*)(Xg + (size_t)pos * HDIM + c4) = sc.u;
}

// ---------------- gate/up GEMM with fused SiLU(g)*u epilogue ----------------
// offsets==nullptr -> shared expert (M = T_TOK, wStride = 0).
// Wave tile: 32(M) x 64(N): A frags a0,a1 share each B frag.

__global__ __launch_bounds__(256) void gateup_gemm(
    const bf16* __restrict__ Aall, const float* __restrict__ Wg,
    const float* __restrict__ Wu, bf16* __restrict__ Out,
    const int* __restrict__ offsets, int K, int N, size_t wStride) {
    __shared__ bf16 As[BM * BKP];
    __shared__ bf16 Bgs[BN * BKP];
    __shared__ bf16 Bus[BN * BKP];

    int e = blockIdx.z;
    int base = 0, Mvalid = T_TOK;
    if (offsets) { base = offsets[e]; Mvalid = offsets[e + 1] - base; }
    int mt = blockIdx.x;
    if (mt * BM >= Mvalid) return;                 // block-uniform
    int n0 = blockIdx.y * BN;

    const bf16*  A  = Aall + (size_t)(base + mt * BM) * K;
    const float* wg = Wg + (size_t)e * wStride;
    const float* wu = Wu + (size_t)e * wStride;

    int tid = threadIdx.x;
    int lane = tid & 31, wave = tid >> 5;
    int wm = wave >> 2;                 // 0..1 : M offset wm*32
    int wn = wave & 3;                  // 0..3 : N offset wn*64

    v8f accG[2][4] = {};
    v8f accU[2][4] = {};

    for (int k0 = 0; k0 < K; k0 += BK) {
        __syncthreads();
        load_A_tile(As, A, K, k0, tid);
        load_B_tile(Bgs, wg, N, k0, n0, tid);
        load_B_tile(Bus, wu, N, k0, n0, tid);
        __syncthreads();
        v16bf a0 = load_frag_A(As + (wm * 32)      * BKP, lane);
        v16bf a1 = load_frag_A(As + (wm * 32 + 16) * BKP, lane);
#pragma unroll
        for (int nt = 0; nt < 4; ++nt) {
            v16bf bg = load_frag_B(Bgs, wn * 64 + nt * 16, lane);
            accG[0][nt] = __builtin_amdgcn_wmma_f32_16x16x32_bf16(
                false, a0, false, bg, (short)0, accG[0][nt], false, false);
            accG[1][nt] = __builtin_amdgcn_wmma_f32_16x16x32_bf16(
                false, a1, false, bg, (short)0, accG[1][nt], false, false);
            v16bf bu = load_frag_B(Bus, wn * 64 + nt * 16, lane);
            accU[0][nt] = __builtin_amdgcn_wmma_f32_16x16x32_bf16(
                false, a0, false, bu, (short)0, accU[0][nt], false, false);
            accU[1][nt] = __builtin_amdgcn_wmma_f32_16x16x32_bf16(
                false, a1, false, bu, (short)0, accU[1][nt], false, false);
        }
    }

    // C/D layout: VGPR r, lanes 0-15 -> M=r, lanes 16-31 -> M=r+8; N = lane&15
    int ncol = lane & 15;
#pragma unroll
    for (int fa = 0; fa < 2; ++fa) {
        int mrow0 = mt * BM + wm * 32 + fa * 16 + (lane >> 4) * 8;
#pragma unroll
        for (int nt = 0; nt < 4; ++nt) {
            int col = n0 + wn * 64 + nt * 16 + ncol;
#pragma unroll
            for (int r = 0; r < 8; ++r) {
                int row = mrow0 + r;
                if (row < Mvalid) {
                    float g = accG[fa][nt][r], u = accU[fa][nt][r];
                    float silu = g * fast_rcp(1.f + __expf(-g));   // v_rcp_f32
                    Out[(size_t)(base + row) * N + col] = (bf16)(silu * u);
                }
            }
        }
    }
}

// ---------------- down GEMM: shared writes, routed scatter-adds -------------

__global__ __launch_bounds__(256) void down_gemm(
    const bf16* __restrict__ Aall, const float* __restrict__ Wd,
    float* __restrict__ out, const int* __restrict__ offsets,
    const int* __restrict__ perm, int K, int N, size_t wStride) {
    __shared__ bf16 As[BM * BKP];
    __shared__ bf16 Bs[BN * BKP];

    int e = blockIdx.z;
    int base = 0, Mvalid = T_TOK;
    if (offsets) { base = offsets[e]; Mvalid = offsets[e + 1] - base; }
    int mt = blockIdx.x;
    if (mt * BM >= Mvalid) return;
    int n0 = blockIdx.y * BN;

    const bf16*  A = Aall + (size_t)(base + mt * BM) * K;
    const float* w = Wd + (size_t)e * wStride;

    int tid = threadIdx.x;
    int lane = tid & 31, wave = tid >> 5;
    int wm = wave >> 2;
    int wn = wave & 3;

    v8f acc[2][4] = {};

    for (int k0 = 0; k0 < K; k0 += BK) {
        __syncthreads();
        load_A_tile(As, A, K, k0, tid);
        load_B_tile(Bs, w, N, k0, n0, tid);
        __syncthreads();
        v16bf a0 = load_frag_A(As + (wm * 32)      * BKP, lane);
        v16bf a1 = load_frag_A(As + (wm * 32 + 16) * BKP, lane);
#pragma unroll
        for (int nt = 0; nt < 4; ++nt) {
            v16bf b = load_frag_B(Bs, wn * 64 + nt * 16, lane);
            acc[0][nt] = __builtin_amdgcn_wmma_f32_16x16x32_bf16(
                false, a0, false, b, (short)0, acc[0][nt], false, false);
            acc[1][nt] = __builtin_amdgcn_wmma_f32_16x16x32_bf16(
                false, a1, false, b, (short)0, acc[1][nt], false, false);
        }
    }

    int ncol = lane & 15;
#pragma unroll
    for (int fa = 0; fa < 2; ++fa) {
        int mrow0 = mt * BM + wm * 32 + fa * 16 + (lane >> 4) * 8;
#pragma unroll
        for (int nt = 0; nt < 4; ++nt) {
            int col = n0 + wn * 64 + nt * 16 + ncol;
#pragma unroll
            for (int r = 0; r < 8; ++r) {
                int row = mrow0 + r;
                if (row < Mvalid) {
                    float v = acc[fa][nt][r];
                    if (perm) {                       // routed: scatter-add
                        int tok = perm[base + row];
                        if (tok >= 0) {
                            float* p = out + (size_t)tok * N + col;
                            *p = *p + v;              // unique (token,col) per lane
                        }
                    } else {                          // shared: plain write
                        out[(size_t)row * N + col] = v;
                    }
                }
            }
        }
    }
}

// ---------------------------------------------------------------------------

extern "C" void kernel_launch(void* const* d_in, const int* in_sizes, int n_in,
                              void* d_out, int out_size, void* d_ws, size_t ws_size,
                              hipStream_t stream) {
    (void)in_sizes; (void)n_in; (void)out_size; (void)ws_size;
    const float* x           = (const float*)d_in[0];
    const float* router_w    = (const float*)d_in[1];
    const float* w_gate      = (const float*)d_in[2];
    const float* w_up        = (const float*)d_in[3];
    const float* w_down      = (const float*)d_in[4];
    const float* shared_gate = (const float*)d_in[5];
    const float* shared_up   = (const float*)d_in[6];
    const float* shared_down = (const float*)d_in[7];
    float* out = (float*)d_out;

    // carve workspace (256B aligned chunks)
    char* p = (char*)d_ws;
    auto carve = [&](size_t bytes) {
        void* r = (void*)p;
        p += (bytes + 255) & ~(size_t)255;
        return r;
    };
    int*   eidx    = (int*)  carve((size_t)T_TOK * 4);
    float* scale   = (float*)carve((size_t)T_TOK * 4);
    int*   counts  = (int*)  carve(NEXP * 4);
    int*   cursor  = (int*)  carve(NEXP * 4);
    int*   offsets = (int*)  carve((NEXP + 1) * 4);
    int*   perm    = (int*)  carve((size_t)ROWS_MAX * 4);
    bf16*  Xs      = (bf16*) carve((size_t)T_TOK * HDIM * 2);
    bf16*  Xg      = (bf16*) carve((size_t)ROWS_MAX * HDIM * 2);
    bf16*  Hbuf    = (bf16*) carve((size_t)ROWS_MAX * IDIM * 2);
    bf16*  SharedH = (bf16*) carve((size_t)T_TOK * IDIM * 2);

    hipMemsetAsync(counts, 0, NEXP * 4, stream);
    hipMemsetAsync(cursor, 0, NEXP * 4, stream);
    hipMemsetAsync(perm, 0xFF, (size_t)ROWS_MAX * 4, stream);          // -1
    hipMemsetAsync(Xg, 0, (size_t)ROWS_MAX * HDIM * 2, stream);        // zero pad rows

    router_kernel<<<T_TOK / 8, 256, 0, stream>>>(x, router_w, eidx, scale, counts);
    offsets_kernel<<<1, 32, 0, stream>>>(counts, offsets);
    scatter_kernel<<<T_TOK, 256, 0, stream>>>(x, eidx, scale, offsets, cursor,
                                              perm, Xg, Xs);

    // shared expert gate/up : [T,H] x [H,I] -> SharedH
    gateup_gemm<<<dim3(T_TOK / BM, IDIM / BN, 1), NTHR, 0, stream>>>(
        Xs, shared_gate, shared_up, SharedH, nullptr, HDIM, IDIM, 0);
    // routed gate/up per expert
    gateup_gemm<<<dim3(T_TOK / BM, IDIM / BN, NEXP), NTHR, 0, stream>>>(
        Xg, w_gate, w_up, Hbuf, offsets, HDIM, IDIM, (size_t)HDIM * IDIM);
    // shared down : writes d_out
    down_gemm<<<dim3(T_TOK / BM, HDIM / BN, 1), NTHR, 0, stream>>>(
        SharedH, shared_down, out, nullptr, nullptr, IDIM, HDIM, 0);
    // routed down : scatter-add into d_out (after shared write, stream-ordered)
    down_gemm<<<dim3(T_TOK / BM, HDIM / BN, NEXP), NTHR, 0, stream>>>(
        Hbuf, w_down, out, offsets, perm, IDIM, HDIM, (size_t)IDIM * HDIM);
}
